// SparseLinear_17927193494282
// MI455X (gfx1250) — matrix operand order repaired and checked by reading, not compiled, not run on hardware
//
#include <hip/hip_runtime.h>
#include <hip/hip_bf16.h>

// ---------------------------------------------------------------------------
// SparseLinear on MI455X (gfx1250): densify COO -> bf16 WMMA GEMM.
//   y[n, r] = sum_k Wd[r, k] * x[n, k] + bias[r]
// M = BATCH = 1024, N = OUT_FEATURES = 1024, K = IN_FEATURES = 1024
//
// GEMM hot loop: double-buffered LDS stages filled with
// GLOBAL_LOAD_ASYNC_TO_LDS_B128 (ASYNCcnt), consumed by
// v_wmma_f32_16x16x32_bf16. Branch-free stage loader (wave-uniform work).
// ---------------------------------------------------------------------------

typedef __attribute__((ext_vector_type(16))) __bf16 v16bf;
typedef __attribute__((ext_vector_type(8)))  __bf16 v8bf;
typedef __attribute__((ext_vector_type(8)))  float  v8f;

typedef __attribute__((address_space(3))) __bf16       lds_bf16;
typedef __attribute__((address_space(1))) const __bf16 glb_bf16;

#define GEMM_M 1024
#define GEMM_N 1024
#define GEMM_K 1024

#define KSTEP    32           // K elements per stage
#define NKB      (GEMM_K / KSTEP)
#define LDSTRIDE 40           // padded row stride in bf16 (80B) -> DS conflict-free

// ---------------------------------------------------------------------------
// Async copy of 16 bytes global -> LDS (per-lane), tracked with ASYNCcnt.
// Inline asm: portable across ROCm7.2 / amdgpu-toolchain builtin arity skew.
// ---------------------------------------------------------------------------
__device__ __forceinline__ void async_copy_b128(const __bf16* gptr, __bf16* sptr) {
    asm volatile("global_load_async_to_lds_b128 %0, %1, off"
                 :
                 : "v"((lds_bf16*)sptr), "v"((glb_bf16*)gptr)
                 : "memory");
}
__device__ __forceinline__ void wait_asynccnt_le5() {
    asm volatile("s_wait_asynccnt 0x5" ::: "memory");
}
__device__ __forceinline__ void wait_asynccnt_0() {
    asm volatile("s_wait_asynccnt 0x0" ::: "memory");
}

// ---------------------------------------------------------------------------
// Kernel 1: zero the dense fp32 weight buffer (1M floats).
// ---------------------------------------------------------------------------
__global__ void sl_zero_kernel(float* __restrict__ Wd) {
    int i = blockIdx.x * blockDim.x + threadIdx.x;
    Wd[i] = 0.0f;
}

// ---------------------------------------------------------------------------
// Kernel 2: scatter-add COO weights into dense W (duplicates accumulate,
// matching jax segment_sum). indices: int64 [2, NNZ] (rows then cols).
// ---------------------------------------------------------------------------
__global__ void sl_scatter_kernel(const long long* __restrict__ idx,
                                  const float* __restrict__ w,
                                  float* __restrict__ Wd, int nnz) {
    int i = blockIdx.x * blockDim.x + threadIdx.x;
    if (i < nnz) {
        int r = (int)idx[i];
        int c = (int)idx[nnz + i];
        atomicAdd(&Wd[r * GEMM_K + c], w[i]);
    }
}

// ---------------------------------------------------------------------------
// Kernel 3: convert x (fp32) and dense W (fp32) to bf16 (1M elements each).
// ---------------------------------------------------------------------------
__global__ void sl_convert_kernel(const float* __restrict__ x,
                                  const float* __restrict__ Wd,
                                  __bf16* __restrict__ Xbf,
                                  __bf16* __restrict__ Wbf) {
    int i = blockIdx.x * blockDim.x + threadIdx.x;
    Xbf[i] = (__bf16)x[i];
    Wbf[i] = (__bf16)Wd[i];
}

// ---------------------------------------------------------------------------
// Kernel 4: bf16 WMMA GEMM, LDS double-buffered via async-to-LDS.
// WG: 256 threads = 8 waves. WG tile 256(M) x 64(N); wave tile 64x32
// (4x2 grid of 16x16 f32 accumulators). Grid: (N/64, M/256) = (16, 4).
//
// Per stage (K slab of 32): sA 256x32 bf16 (rows padded to 40), sB 64x32.
// Each thread issues exactly 5 async b128 copies per stage (4 A + 1 B),
// branch-free. Pipeline: issue(s+1); s_wait_asynccnt<=5 (stage s complete,
// ASYNCcnt retires loads in order); barrier; 8 WMMAs from LDS; barrier.
//
// Fragment layouts used here, contiguous K (not the ISA table order) (16-bit, wave32):
//   A (16x32): lane&15 = M row; half-wave0 K0..15, half-wave1
//              K16..31 -> two 16B DS reads.
//   B (32x16): lane&15 = N col; half-wave0 K0..15, half-wave1 K16..31
//              -> one 32B DS read (B[k][n] = Wd[n][k], K-contiguous).
//   C/D: vgpr r -> M = r + 8*(lane>=16), N = lane&15.
// ---------------------------------------------------------------------------
__global__ void __launch_bounds__(256)
sl_gemm_kernel(const __bf16* __restrict__ A,   // Xbf  [M x K] row-major
               const __bf16* __restrict__ Bw,  // Wbf  [N x K] row-major
               const float* __restrict__ bias, // [N]
               float* __restrict__ out) {      // [M x N] row-major
    __shared__ __bf16 sA[2][256 * LDSTRIDE];   // 2 x 20 KB
    __shared__ __bf16 sB[2][64 * LDSTRIDE];    // 2 x  5 KB

    const int tid    = threadIdx.x;
    const int lane   = tid & 31;
    const int waveId = tid >> 5;
    const int wm     = waveId & 3;   // 4 waves along M
    const int wn     = waveId >> 2;  // 2 waves along N

    const int m_wg = blockIdx.y * 256;
    const int n_wg = blockIdx.x * 64;

    const int lmod = lane & 15;
    const int lhi  = lane >> 4;      // 0: lanes 0-15, 1: lanes 16-31

    // Branch-free B-loader coordinates: 256 threads cover 64 rows x 32 elems.
    //   threads 0-127  -> rows 0..63, K elems {0..7 | 8..15}
    //   threads 128-255-> rows 0..63, K elems {16..23 | 24..31}
    const int bhalf = tid >> 7;              // 0 or 1 (wave-uniform)
    const int bidx  = tid & 127;
    const int brow  = bidx >> 1;             // 0..63
    const int boff  = (bidx & 1) * 8 + bhalf * 16;  // 0,8,16,24

    v8f acc[4][2] = {};

    // ---- async stage loader: global slab (k..k+31) -> LDS stage s ----
    auto load_stage = [&](int s, int k) {
        // A: 1024 chunks of 16B; thread t handles chunks {t, t+256, ...}
        #pragma unroll
        for (int c = 0; c < 4; ++c) {
            const int chunk = c * 256 + tid;
            const int row   = chunk >> 2;            // 0..255
            const int off   = (chunk & 3) * 8;       // 0,8,16,24 (bf16 elems)
            async_copy_b128(A + (m_wg + row) * GEMM_K + k + off,
                            &sA[s][row * LDSTRIDE + off]);
        }
        // B: 256 chunks of 16B; one per thread, branch-free
        async_copy_b128(Bw + (n_wg + brow) * GEMM_K + k + boff,
                        &sB[s][brow * LDSTRIDE + boff]);
    };

    load_stage(0, 0);

    for (int kb = 0; kb < NKB; ++kb) {
        const int s = kb & 1;
        if (kb + 1 < NKB) {
            load_stage(s ^ 1, (kb + 1) * KSTEP);
            wait_asynccnt_le5();   // first 5 (stage s) retired, in-order
        } else {
            wait_asynccnt_0();
        }
        __syncthreads();           // all waves' stage-s data visible

        // ---- fragments from LDS ----
        v16bf afr[4];
        #pragma unroll
        for (int mt = 0; mt < 4; ++mt) {
            const __bf16* p = &sA[s][(wm * 64 + mt * 16 + lmod) * LDSTRIDE + lhi * 16];
            v8bf lo = *(const v8bf*)(p);        // K ka..ka+7
            v8bf hi = *(const v8bf*)(p + 8);    // K ka+8..ka+15
            afr[mt] = __builtin_shufflevector(lo, hi,
                0, 1, 2, 3, 4, 5, 6, 7, 8, 9, 10, 11, 12, 13, 14, 15);
        }
        v16bf bfr[2];
        #pragma unroll
        for (int nt = 0; nt < 2; ++nt) {
            const __bf16* p = &sB[s][(wn * 32 + nt * 16 + lmod) * LDSTRIDE + lhi * 16];
            bfr[nt] = *(const v16bf*)(p);       // 16 contiguous K values
        }

        // ---- 8 WMMA ops: D = A x B + C ----
        #pragma unroll
        for (int mt = 0; mt < 4; ++mt) {
            #pragma unroll
            for (int nt = 0; nt < 2; ++nt) {
                acc[mt][nt] = __builtin_amdgcn_wmma_f32_16x16x32_bf16(
                    false, afr[mt], false, bfr[nt],
                    (short)0, acc[mt][nt], false, false);
            }
        }
        __syncthreads();           // stage s free for the next async fill
    }

    // ---- epilogue: bias add + store (C/D layout) ----
    #pragma unroll
    for (int nt = 0; nt < 2; ++nt) {
        const int col = n_wg + wn * 32 + nt * 16 + lmod;
        const float bv = bias[col];
        #pragma unroll
        for (int mt = 0; mt < 4; ++mt) {
            const int row0 = m_wg + wm * 64 + mt * 16 + (lhi << 3);
            #pragma unroll
            for (int r = 0; r < 8; ++r) {
                out[(row0 + r) * GEMM_N + col] = acc[mt][nt][r] + bv;
            }
        }
    }
}

// ---------------------------------------------------------------------------
// Launch: zero -> scatter -> convert -> WMMA GEMM (all on `stream`).
// Workspace: [0,4MB) Wd fp32 | [4MB,6MB) Wbf bf16 | [6MB,8MB) Xbf bf16
// ---------------------------------------------------------------------------
extern "C" void kernel_launch(void* const* d_in, const int* in_sizes, int n_in,
                              void* d_out, int out_size, void* d_ws, size_t ws_size,
                              hipStream_t stream) {
    const float*     x    = (const float*)d_in[0];
    const float*     w    = (const float*)d_in[1];
    const float*     bias = (const float*)d_in[2];
    const long long* idx  = (const long long*)d_in[3];
    const int nnz = in_sizes[1];

    char* ws = (char*)d_ws;
    float*  Wd  = (float*)ws;                          // 1M fp32  (4 MB)
    __bf16* Wbf = (__bf16*)(ws + (4u << 20));          // 1M bf16  (2 MB)
    __bf16* Xbf = (__bf16*)(ws + (6u << 20));          // 1M bf16  (2 MB)
    float*  out = (float*)d_out;

    const int total = GEMM_N * GEMM_K;                 // 1M elements

    sl_zero_kernel<<<total / 256, 256, 0, stream>>>(Wd);
    sl_scatter_kernel<<<(nnz + 255) / 256, 256, 0, stream>>>(idx, w, Wd, nnz);
    sl_convert_kernel<<<total / 256, 256, 0, stream>>>(x, Wd, Xbf, Wbf);

    dim3 grid(GEMM_N / 64, GEMM_M / 256);              // (16, 4)
    sl_gemm_kernel<<<grid, 256, 0, stream>>>(Xbf, Wbf, bias, out);
}
